// NEHDLayer_16045997817888
// MI455X (gfx1250) — matrix operand order repaired and checked by reading, not compiled, not run on hardware
//
#include <hip/hip_runtime.h>
#include <hip/hip_bf16.h>

// NEHD layer, fully fused for MI455X (gfx1250, wave32, WMMA).
// Edge conv: v_wmma_f32_16x16x32_f16, 32 pixels x 8 channels per WMMA via
//   pixel-pairing (rows 0-7 = weights at K0-7,K16, rows 8-15 = same weights
//   duplicated at K8-15,K24, so B columns carry two independent pixels).
// No-edge conv + sigmoid + RBF histogram + straddling-group normalization on
// VALU over LDS-resident xx; deterministic two-stage reduction -> d_out.

typedef __attribute__((ext_vector_type(16))) _Float16 v16h;
typedef __attribute__((ext_vector_type(8)))  float    v8f;

#define TC   3            // input channels
#define TNB  8            // bins per channel
#define BINN 27           // C*NB + C
#define TILE 16           // output tile (per dim)
#define XT   20           // x halo tile  = TILE + 4
#define XXT  18           // xx halo tile = TILE + 2
#define NPIX (XXT * XXT)  // 324 xx pixels per group
#define NBLK ((NPIX + 31) / 32)            // 11 pixel-pair blocks per group
#define NTILES 32         // ceil(510/16) -> 32x32 tiles cover [0,512)
#define HW_OUT 510

__global__ __launch_bounds__(256)
void nehd_fused(const float* __restrict__ x,
                const float* __restrict__ edge_w,    // [24][9]
                const float* __restrict__ no_edge_w, // [3][8][9]
                const float* __restrict__ centers,   // [27]
                const float* __restrict__ widths,    // [27]
                float* __restrict__ ws)              // [N][1024][27] partials
{
    __shared__ _Float16 s_xh[TC][XT][XT];      // 2.4 KB, f16 staging for WMMA
    __shared__ float s_xx[TC][NPIX][TNB];      // 31.1 KB (f32 edge responses)
    __shared__ float s_ew[TC * TNB][9];        // edge weights
    __shared__ float s_nw[TC][TNB][9];         // no-edge weights
    __shared__ float s_cent[BINN], s_wid[BINN];
    __shared__ float s_wred[8][BINN];          // per-wave partials

    const int tid  = threadIdx.x;
    const int lane = tid & 31;
    const int wave = tid >> 5;
    const int txb  = blockIdx.x, tyb = blockIdx.y, n = blockIdx.z;
    const int ox0  = txb * TILE, oy0 = tyb * TILE;  // output-tile origin

    // ---- stage 0: params + x halo tile (converted to f16) into LDS -------
    for (int i = tid; i < TC * TNB * 9; i += 256) s_ew[i / 9][i % 9] = edge_w[i];
    for (int i = tid; i < TC * TNB * 9; i += 256) ((float*)s_nw)[i] = no_edge_w[i];
    if (tid < BINN) { s_cent[tid] = centers[tid]; s_wid[tid] = widths[tid]; }

    for (int i = tid; i < TC * XT * XT; i += 256) {
        const int c = i / (XT * XT);
        const int r = (i / XT) % XT;
        const int cc = i % XT;
        const int gy = oy0 - 1 + r, gx = ox0 - 1 + cc;
        float v = 0.f;
        if (gy >= 0 && gy < 512 && gx >= 0 && gx < 512)
            v = x[(((size_t)n * TC + c) * 512 + gy) * 512 + gx];
        s_xh[c][r][cc] = (_Float16)v;
    }
    __syncthreads();

    // ---- stage 1+2: edge conv via WMMA -> s_xx ---------------------------
    // A (16-bit 16x32 layout): lane L<16 holds row M=L, halves0-7 = K0-7,
    // halves8-15 = K16-23; lane L>=16 holds row M=L-16, halves0-7 = K8-15,
    // halves8-15 = K24-31. Rows 0-7: taps 0-8 at K0-7,K16 (lanes 0-7).
    // Rows 8-15: SAME weights at K8-15,K24 (lanes 24-31), matching B.
    // B: every lane's halves0-8 are the 9 taps of pixel p = blk*32 + lane
    //    (lanes 0-15 land in K0-7,K16 for columns, lanes 16-31 in K8-15,K24).
    // C/D: c[r] on lane L = channel r of pixel blk*32 + L, for ALL lanes.
    #pragma unroll
    for (int g = 0; g < TC; ++g) {
        v16h a = {};
        {
            const int m = lane & 15;
            const float* w = s_ew[g * 8 + (m & 7)];
            if ((lane < 16) == (m < 8)) {
                #pragma unroll
                for (int t = 0; t < 9; ++t) a[t] = (_Float16)w[t];
            }
        }
        for (int blk = wave; blk < NBLK; blk += 8) {
            const int p = blk * 32 + lane;     // one pixel per lane
            const bool pv = p < NPIX;
            v16h b = {};
            if (pv) {
                const int r  = p / XXT, cc = p % XXT;
                const int gy = oy0 - 1 + r, gx = ox0 - 1 + cc; // global xx coord
                if (gy >= 0 && gy < HW_OUT && gx >= 0 && gx < HW_OUT) {
                    #pragma unroll
                    for (int t = 0; t < 9; ++t)
                        b[t] = s_xh[g][r + t / 3][cc + t % 3];
                } // else taps stay 0 -> SAME zero-pad of xx
            }
            v8f cacc = {};
            cacc = __builtin_amdgcn_wmma_f32_16x16x32_f16(
                false, a, false, b, (short)0, cacc, false, false);
            if (pv) {
                float4* dst = (float4*)s_xx[g][p];
                dst[0] = make_float4(cacc[0], cacc[1], cacc[2], cacc[3]);
                dst[1] = make_float4(cacc[4], cacc[5], cacc[6], cacc[7]);
            }
        }
    }
    __syncthreads();

    // ---- stage 3: no-edge conv + RBF histogram + normalization -----------
    float acc[BINN];
    #pragma unroll
    for (int ch = 0; ch < BINN; ++ch) acc[ch] = 0.f;

    const int oy = tid / TILE, ox = tid % TILE;   // one output pixel / thread
    const int gy = oy0 + oy, gx = ox0 + ox;
    if (gy < HW_OUT && gx < HW_OUT) {
        float vals[BINN];
        // no-edge: 3x3 SAME over the 8 bins of each group, then sigmoid
        #pragma unroll
        for (int c = 0; c < TC; ++c) {
            float s = 0.f;
            #pragma unroll
            for (int t = 0; t < 9; ++t) {
                const float* xv = s_xx[c][(oy + t / 3) * XXT + (ox + t % 3)];
                #pragma unroll
                for (int bb = 0; bb < TNB; ++bb)
                    s += xv[bb] * s_nw[c][bb][t];
            }
            vals[24 + c] = 1.f / (1.f + __expf(-s));
        }
        // edge bins at the output pixel (xx local center = (oy+1, ox+1))
        #pragma unroll
        for (int ch = 0; ch < 24; ++ch)
            vals[ch] = s_xx[ch >> 3][(oy + 1) * XXT + (ox + 1)][ch & 7];
        // RBF membership
        float h[BINN];
        #pragma unroll
        for (int ch = 0; ch < BINN; ++ch) {
            const float z = (vals[ch] + s_cent[ch]) * s_wid[ch];
            h[ch] = __expf(-(z * z));
        }
        // normalization groups straddle channels: [0..8],[9..17],[18..26]
        #pragma unroll
        for (int g = 0; g < TC; ++g) {
            float s = 0.f;
            #pragma unroll
            for (int k = 0; k < 9; ++k) s += h[g * 9 + k];
            const float inv = 1.f / (s + 1e-5f);
            #pragma unroll
            for (int k = 0; k < 9; ++k) acc[g * 9 + k] = h[g * 9 + k] * inv;
        }
    }

    // ---- stage 4: wave32 shuffle reduce, then cross-wave via LDS ---------
    #pragma unroll
    for (int ch = 0; ch < BINN; ++ch) {
        float v = acc[ch];
        #pragma unroll
        for (int off = 16; off > 0; off >>= 1) v += __shfl_xor(v, off, 32);
        if (lane == 0) s_wred[wave][ch] = v;
    }
    __syncthreads();
    if (tid < BINN) {
        float s = 0.f;
        #pragma unroll
        for (int w = 0; w < 8; ++w) s += s_wred[w][tid];
        const int tile = tyb * NTILES + txb;
        ws[((size_t)n * (NTILES * NTILES) + tile) * BINN + tid] = s;
    }
}

// Deterministic fixed-order reduction of tile partials + mean scaling.
__global__ __launch_bounds__(64)
void nehd_reduce(const float* __restrict__ ws, float* __restrict__ out)
{
    const int idx = blockIdx.x * 64 + threadIdx.x;   // [0, 16*27)
    if (idx >= 16 * BINN) return;
    const int n = idx / BINN, ch = idx % BINN;
    float s = 0.f;
    const int ntiles = NTILES * NTILES;
    for (int t = 0; t < ntiles; ++t)
        s += ws[((size_t)n * ntiles + t) * BINN + ch];
    out[idx] = s * (1.f / (510.f * 510.f));
}

extern "C" void kernel_launch(void* const* d_in, const int* in_sizes, int n_in,
                              void* d_out, int out_size, void* d_ws, size_t ws_size,
                              hipStream_t stream) {
    const float* x         = (const float*)d_in[0];  // [16,3,512,512]
    const float* edge_w    = (const float*)d_in[1];  // [24,1,3,3]
    const float* no_edge_w = (const float*)d_in[2];  // [3,8,3,3]
    const float* centers   = (const float*)d_in[3];  // [27]
    const float* widths    = (const float*)d_in[4];  // [27]
    float* out = (float*)d_out;                      // [16,27,1,1] = 432
    float* ws  = (float*)d_ws;                       // 16*1024*27*4 = 1.7 MB

    dim3 grid(NTILES, NTILES, 16);
    nehd_fused<<<grid, 256, 0, stream>>>(x, edge_w, no_edge_w, centers, widths, ws);
    nehd_reduce<<<dim3((16 * BINN + 63) / 64), 64, 0, stream>>>(ws, out);
}
